// SuperpositionAttention_8555574854240
// MI455X (gfx1250) — matrix operand order repaired and checked by reading, not compiled, not run on hardware
//
#include <hip/hip_runtime.h>

typedef __bf16 bf16;
typedef __attribute__((ext_vector_type(16))) __bf16 v16bf;
typedef __attribute__((ext_vector_type(8)))  __bf16 v8bf;
typedef __attribute__((ext_vector_type(8)))  float  v8f;
typedef __attribute__((ext_vector_type(4)))  int    v4i;

#define WMMA_BF16(a, b, c) \
  __builtin_amdgcn_wmma_f32_16x16x32_bf16(false, (a), false, (b), (short)0, (c), false, false)

// ---- CDNA5 async global->LDS copy (ASYNCcnt path), with safe fallback -----
#if defined(__has_builtin)
#if __has_builtin(__builtin_amdgcn_global_load_async_to_lds_b128)
#define HAVE_ASYNC_LDS 1
#endif
#if __has_builtin(__builtin_amdgcn_s_wait_asynccnt)
#define HAVE_WAIT_ASYNC_BUILTIN 1
#endif
#endif
#ifndef HAVE_ASYNC_LDS
#define HAVE_ASYNC_LDS 0
#endif
#ifndef HAVE_WAIT_ASYNC_BUILTIN
#define HAVE_WAIT_ASYNC_BUILTIN 0
#endif

// builtin signature (from the compiler diagnostic): v4i ptrs, global + LDS AS
typedef __attribute__((address_space(1))) v4i* p1v4i;
typedef __attribute__((address_space(3))) v4i* p3v4i;

static __device__ inline void async_copy16(const bf16* g, bf16* l) {
#if HAVE_ASYNC_LDS
  __builtin_amdgcn_global_load_async_to_lds_b128((p1v4i)g, (p3v4i)l, 0, 0);
#else
  *(v8bf*)l = *(const v8bf*)g;
#endif
}

static __device__ inline void wait_async() {
#if HAVE_ASYNC_LDS
#if HAVE_WAIT_ASYNC_BUILTIN
  __builtin_amdgcn_s_wait_asynccnt(0);
#else
  asm volatile("s_wait_asynccnt 0" ::: "memory");
#endif
#endif
}

// A-operand fragment, 16x32 bf16. Row = M (lane&15). Per ISA table:
// lanes 0-15: elems 0..7 -> K=0..7, elems 8..15 -> K=16..23
// lanes16-31: elems 0..7 -> K=8..15, elems 8..15 -> K=24..31
static __device__ inline v16bf make_afrag(const bf16* row, int hi8) {
  v8bf lo = *(const v8bf*)(row + hi8);
  v8bf hi = *(const v8bf*)(row + 16 + hi8);
  v16bf r;
#pragma unroll
  for (int i = 0; i < 8; ++i) { r[i] = lo[i]; r[i + 8] = hi[i]; }
  return r;
}

// B-operand fragment, 32x16 bf16, stored N-major (contraction contiguous).
// lanes 0-15 (N=lane): K=0..7,16..23; lanes 16-31 (N=lane-16): K=8..15,24..31.
static __device__ inline v16bf make_bfrag(const bf16* row, int hi16) {
  v8bf lo = *(const v8bf*)(row + (hi16 >> 1));
  v8bf hi = *(const v8bf*)(row + 16 + (hi16 >> 1));
  v16bf r;
#pragma unroll
  for (int i = 0; i < 8; ++i) { r[i] = lo[i]; r[i + 8] = hi[i]; }
  return r;
}

// Stage one 128x32 bf16 tile (rows row0.., cols k0..k0+31) from a row-major
// source with stride 512 into padded LDS, as 512 16-byte async transfers.
static __device__ inline void stage_tile(const bf16* __restrict__ src, int row0,
                                         int k0, bf16 (*dst)[40], int tid) {
#pragma unroll
  for (int i = 0; i < 2; ++i) {
    int chunk = tid + i * 256;             // 0..511
    int r = chunk >> 2, p = (chunk & 3) * 8;
    async_copy16(src + (size_t)(row0 + r) * 512 + k0 + p, &dst[r][p]);
  }
}

// ---------------------------------------------------------------------------
// Prep kernels: x -> bf16 row-major; weights -> bf16, N-major (transposed),
// keeping only the first 512 output columns of Wq/Wk/Wv (heads 4..7 are dead
// code in the reference because of out[..., :D]).
// ---------------------------------------------------------------------------
__global__ __launch_bounds__(256) void prep_x_kernel(const float* __restrict__ x,
                                                     bf16* __restrict__ xb) {
  size_t i = (size_t)blockIdx.x * 256 + threadIdx.x;  // 4096*512 elems
  xb[i] = (bf16)x[i];
}

__global__ __launch_bounds__(256) void prep_w_kernel(
    const float* __restrict__ Wq, const float* __restrict__ Wk,
    const float* __restrict__ Wv, const float* __restrict__ Wo,
    bf16* __restrict__ Wtq, bf16* __restrict__ Wtk, bf16* __restrict__ Wtv,
    bf16* __restrict__ Wot) {
  size_t i = (size_t)blockIdx.x * 256 + threadIdx.x;  // 512*512 elems
  int n = (int)(i >> 9), k = (int)(i & 511);
  Wtq[i] = (bf16)Wq[(size_t)k * 1024 + n];
  Wtk[i] = (bf16)Wk[(size_t)k * 1024 + n];
  Wtv[i] = (bf16)Wv[(size_t)k * 1024 + n];
  Wot[i] = (bf16)Wo[(size_t)k * 512 + n];
}

// ---------------------------------------------------------------------------
// Kernel 1: fused QKV projection (heads 0..3 only).
//   C(4096x512) = xb @ Wt^T + bias, double-buffered async-LDS staging,
//   bf16 WMMA, f32 accum. Outputs head-major bf16: Q[bh][s][128],
//   K[bh][s][128], Krot = [-ki, kr], Vt[bh][d][s].
// ---------------------------------------------------------------------------
__global__ __launch_bounds__(256) void qkv_kernel(
    const bf16* __restrict__ xb,
    const bf16* __restrict__ Wtq, const bf16* __restrict__ Wtk,
    const bf16* __restrict__ Wtv,
    const float* __restrict__ bq, const float* __restrict__ bk,
    const float* __restrict__ bv,
    bf16* __restrict__ Qb, bf16* __restrict__ Kb,
    bf16* __restrict__ Krot, bf16* __restrict__ Vt) {
  __shared__ bf16 As[2][128][40];
  __shared__ bf16 Bs[2][128][40];

  const int tid  = threadIdx.x;
  const int wid  = tid >> 5, lane = tid & 31;
  const int lrow = lane & 15;
  const int hi8  = (lane >= 16) ? 8 : 0;
  const int hi16 = (lane >= 16) ? 16 : 0;
  const int wm = wid >> 1, wn = wid & 1;
  const int m0 = blockIdx.x * 128;
  const int n0 = blockIdx.y * 128;
  const int mode = blockIdx.z;  // 0=Q, 1=K, 2=V

  const bf16* Wt    = (mode == 0) ? Wtq : (mode == 1) ? Wtk : Wtv;
  const float* bias = (mode == 0) ? bq : (mode == 1) ? bk : bv;

  v8f acc[2][4];
#pragma unroll
  for (int i = 0; i < 2; ++i)
#pragma unroll
    for (int j = 0; j < 4; ++j)
#pragma unroll
      for (int e = 0; e < 8; ++e) acc[i][j][e] = 0.0f;

  stage_tile(xb, m0, 0, As[0], tid);
  stage_tile(Wt, n0, 0, Bs[0], tid);

  for (int kt = 0; kt < 16; ++kt) {
    wait_async();
    __syncthreads();
    if (kt + 1 < 16) {  // overlap next tile's DMA with current WMMAs
      stage_tile(xb, m0, (kt + 1) * 32, As[(kt + 1) & 1], tid);
      stage_tile(Wt, n0, (kt + 1) * 32, Bs[(kt + 1) & 1], tid);
    }
    const bf16 (*A)[40]  = As[kt & 1];
    const bf16 (*Bt)[40] = Bs[kt & 1];

    v16bf af[2];
#pragma unroll
    for (int mt = 0; mt < 2; ++mt)
      af[mt] = make_afrag(&A[wm * 32 + mt * 16 + lrow][0], hi8);
#pragma unroll
    for (int nt = 0; nt < 4; ++nt) {
      v16bf bfr = make_bfrag(&Bt[wn * 64 + nt * 16 + lrow][0], hi16);
#pragma unroll
      for (int mt = 0; mt < 2; ++mt)
        acc[mt][nt] = WMMA_BF16(af[mt], bfr, acc[mt][nt]);
    }
  }

#pragma unroll
  for (int mt = 0; mt < 2; ++mt)
#pragma unroll
    for (int nt = 0; nt < 4; ++nt)
#pragma unroll
      for (int r = 0; r < 8; ++r) {
        int M = m0 + wm * 32 + mt * 16 + r + hi8;
        int N = n0 + wn * 64 + nt * 16 + lrow;
        float c = acc[mt][nt][r] + bias[N];
        int b = M >> 11, s = M & 2047;
        int h = N >> 7,  j = N & 127;
        size_t idx = (((size_t)(b * 4 + h)) * 2048 + s) * 128 + j;
        if (mode == 0) {
          Qb[idx] = (bf16)c;
        } else if (mode == 1) {
          Kb[idx] = (bf16)c;
          size_t base = (((size_t)(b * 4 + h)) * 2048 + s) * 128;
          if (j < 64) Krot[base + j + 64] = (bf16)c;     // kr -> slot 64..127
          else        Krot[base + j - 64] = (bf16)(-c);  // ki -> -ki in 0..63
        } else {
          Vt[(((size_t)(b * 4 + h)) * 128 + j) * 2048 + s] = (bf16)c;
        }
      }
}

// ---------------------------------------------------------------------------
// Kernel 2: flash-style superposition attention for one (b, h<4, 128 q rows).
//   re = Q128·K128^T, im = Q128·Krot^T, score = (re^2+im^2)*SCALE^2/TEMP,
//   online softmax over key blocks of 32, O += P @ V128.
// Each wave owns 16 q rows; Q fragments + O accumulators live in VGPRs.
// ---------------------------------------------------------------------------
__global__ __launch_bounds__(256) void attn_kernel(
    const bf16* __restrict__ Q, const bf16* __restrict__ K,
    const bf16* __restrict__ Krot, const bf16* __restrict__ Vt,
    bf16* __restrict__ Attn) {
  __shared__ bf16 Pbuf[8][16][40];  // per-wave P transpose staging

  const int tid  = threadIdx.x;
  const int wid  = tid >> 5, lane = tid & 31;
  const int lrow = lane & 15;
  const int hi8  = (lane >= 16) ? 8 : 0;   // also the C-layout row offset
  const int hi16 = (lane >= 16) ? 16 : 0;
  const int bh = blockIdx.y;               // b*4 + h, h in 0..3
  const int q0 = blockIdx.x * 128 + wid * 16;

  const float alpha = 0.03125f;  // (1/sqrt(64))^2 / 0.5

  const bf16* Qbase  = Q    + (size_t)bh * 2048 * 128;
  const bf16* Kbase  = K    + (size_t)bh * 2048 * 128;
  const bf16* Krbase = Krot + (size_t)bh * 2048 * 128;
  const bf16* Vbase  = Vt   + (size_t)bh * 128 * 2048;

  v16bf qf[4];
#pragma unroll
  for (int kk = 0; kk < 4; ++kk)
    qf[kk] = make_afrag(Qbase + (size_t)(q0 + lrow) * 128 + kk * 32, hi8);

  v8f o[8];
  float rmax[8], rsum[8];
#pragma unroll
  for (int dt = 0; dt < 8; ++dt)
#pragma unroll
    for (int e = 0; e < 8; ++e) o[dt][e] = 0.0f;
#pragma unroll
  for (int r = 0; r < 8; ++r) { rmax[r] = -1e30f; rsum[r] = 0.0f; }

  for (int kb = 0; kb < 2048; kb += 32) {
    // prefetch next key block (hint; hides L2 latency behind softmax VALU)
    if (kb + 32 < 2048) {
      __builtin_prefetch(Kbase  + (size_t)(kb + 32 + lane) * 128, 0, 1);
      __builtin_prefetch(Krbase + (size_t)(kb + 32 + lane) * 128, 0, 1);
      __builtin_prefetch(Vbase + (size_t)(lane * 4) * 2048 + kb + 32, 0, 1);
    }

    v8f sre[2], sim[2];
#pragma unroll
    for (int nt = 0; nt < 2; ++nt)
#pragma unroll
      for (int e = 0; e < 8; ++e) { sre[nt][e] = 0.0f; sim[nt][e] = 0.0f; }

#pragma unroll
    for (int nt = 0; nt < 2; ++nt) {
      const bf16* kr  = Kbase  + (size_t)(kb + nt * 16 + lrow) * 128;
      const bf16* kr2 = Krbase + (size_t)(kb + nt * 16 + lrow) * 128;
#pragma unroll
      for (int kk = 0; kk < 4; ++kk)
        sre[nt] = WMMA_BF16(qf[kk], make_bfrag(kr + kk * 32, hi16), sre[nt]);
#pragma unroll
      for (int kk = 0; kk < 4; ++kk)
        sim[nt] = WMMA_BF16(qf[kk], make_bfrag(kr2 + kk * 32, hi16), sim[nt]);
    }

    // scores + online softmax (row stats replicated across 16-lane halves)
    float sc0[8], sc1[8], mb[8];
#pragma unroll
    for (int r = 0; r < 8; ++r) {
      float a0 = (sre[0][r] * sre[0][r] + sim[0][r] * sim[0][r]) * alpha;
      float a1 = (sre[1][r] * sre[1][r] + sim[1][r] * sim[1][r]) * alpha;
      sc0[r] = a0; sc1[r] = a1;
      float m = fmaxf(a0, a1);
      m = fmaxf(m, __shfl_xor(m, 1));
      m = fmaxf(m, __shfl_xor(m, 2));
      m = fmaxf(m, __shfl_xor(m, 4));
      m = fmaxf(m, __shfl_xor(m, 8));
      mb[r] = m;
    }
#pragma unroll
    for (int r = 0; r < 8; ++r) {
      float mnew = fmaxf(rmax[r], mb[r]);
      float corr = __expf(rmax[r] - mnew);
      rmax[r] = mnew;
      float p0 = __expf(sc0[r] - mnew);
      float p1 = __expf(sc1[r] - mnew);
      float t = p0 + p1;
      t += __shfl_xor(t, 1);
      t += __shfl_xor(t, 2);
      t += __shfl_xor(t, 4);
      t += __shfl_xor(t, 8);
      rsum[r] = rsum[r] * corr + t;
#pragma unroll
      for (int dt = 0; dt < 8; ++dt) o[dt][r] *= corr;
      Pbuf[wid][r + hi8][lrow]      = (bf16)p0;  // C-layout -> row-major P
      Pbuf[wid][r + hi8][16 + lrow] = (bf16)p1;
    }

    // P (16x32) as A-fragment, V chunks as B-fragments: O += P @ V128
    v16bf pf = make_afrag(&Pbuf[wid][lrow][0], hi8);
#pragma unroll
    for (int dt = 0; dt < 8; ++dt) {
      const bf16* vrow = Vbase + (size_t)(dt * 16 + lrow) * 2048 + kb;
      o[dt] = WMMA_BF16(pf, make_bfrag(vrow, hi16), o[dt]);
    }
  }

  // normalize and store [real(64) | imag(64)] into channels h*128 + d
  const int b = bh >> 2, h = bh & 3;
#pragma unroll
  for (int dt = 0; dt < 8; ++dt)
#pragma unroll
    for (int r = 0; r < 8; ++r) {
      int s = q0 + r + hi8;
      float val = o[dt][r] / rsum[r];
      Attn[((size_t)b * 2048 + s) * 512 + h * 128 + dt * 16 + lrow] = (bf16)val;
    }
}

// ---------------------------------------------------------------------------
// Kernel 3: output projection, out(4096x512) = Attn(bf16) @ Wot^T + bo, f32.
// ---------------------------------------------------------------------------
__global__ __launch_bounds__(256) void oproj_kernel(
    const bf16* __restrict__ Ab, const bf16* __restrict__ Wot,
    const float* __restrict__ bo, float* __restrict__ out) {
  __shared__ bf16 As[2][128][40];
  __shared__ bf16 Bs[2][128][40];

  const int tid  = threadIdx.x;
  const int wid  = tid >> 5, lane = tid & 31;
  const int lrow = lane & 15;
  const int hi8  = (lane >= 16) ? 8 : 0;
  const int hi16 = (lane >= 16) ? 16 : 0;
  const int wm = wid >> 1, wn = wid & 1;
  const int m0 = blockIdx.x * 128;
  const int n0 = blockIdx.y * 128;

  v8f acc[2][4];
#pragma unroll
  for (int i = 0; i < 2; ++i)
#pragma unroll
    for (int j = 0; j < 4; ++j)
#pragma unroll
      for (int e = 0; e < 8; ++e) acc[i][j][e] = 0.0f;

  stage_tile(Ab, m0, 0, As[0], tid);
  stage_tile(Wot, n0, 0, Bs[0], tid);

  for (int kt = 0; kt < 16; ++kt) {
    wait_async();
    __syncthreads();
    if (kt + 1 < 16) {
      stage_tile(Ab, m0, (kt + 1) * 32, As[(kt + 1) & 1], tid);
      stage_tile(Wot, n0, (kt + 1) * 32, Bs[(kt + 1) & 1], tid);
    }
    const bf16 (*A)[40]  = As[kt & 1];
    const bf16 (*Bt)[40] = Bs[kt & 1];

    v16bf af[2];
#pragma unroll
    for (int mt = 0; mt < 2; ++mt)
      af[mt] = make_afrag(&A[wm * 32 + mt * 16 + lrow][0], hi8);
#pragma unroll
    for (int nt = 0; nt < 4; ++nt) {
      v16bf bfr = make_bfrag(&Bt[wn * 64 + nt * 16 + lrow][0], hi16);
#pragma unroll
      for (int mt = 0; mt < 2; ++mt)
        acc[mt][nt] = WMMA_BF16(af[mt], bfr, acc[mt][nt]);
    }
  }

#pragma unroll
  for (int mt = 0; mt < 2; ++mt)
#pragma unroll
    for (int nt = 0; nt < 4; ++nt)
#pragma unroll
      for (int r = 0; r < 8; ++r) {
        int M = m0 + wm * 32 + mt * 16 + r + hi8;
        int N = n0 + wn * 64 + nt * 16 + lrow;
        out[(size_t)M * 512 + N] = acc[mt][nt][r] + bo[N];
      }
}

extern "C" void kernel_launch(void* const* d_in, const int* in_sizes, int n_in,
                              void* d_out, int out_size, void* d_ws, size_t ws_size,
                              hipStream_t stream) {
  (void)in_sizes; (void)n_in; (void)out_size; (void)ws_size;
  const float* x  = (const float*)d_in[0];
  const float* Wq = (const float*)d_in[1];
  const float* bq = (const float*)d_in[2];
  const float* Wk = (const float*)d_in[3];
  const float* bk = (const float*)d_in[4];
  const float* Wv = (const float*)d_in[5];
  const float* bv = (const float*)d_in[6];
  const float* Wo = (const float*)d_in[7];
  const float* bo = (const float*)d_in[8];
  float* out = (float*)d_out;

  char* w = (char*)d_ws;
  const size_t MB = 1024 * 1024;
  bf16* Qb   = (bf16*)(w);            // 4 MB  (2*4*2048*128 bf16)
  bf16* Kb   = (bf16*)(w + 4 * MB);   // 4 MB
  bf16* Krot = (bf16*)(w + 8 * MB);   // 4 MB
  bf16* Vt   = (bf16*)(w + 12 * MB);  // 4 MB
  bf16* Attn = (bf16*)(w + 16 * MB);  // 4 MB  (2*2048*512 bf16)
  bf16* xb   = (bf16*)(w + 20 * MB);  // 4 MB  (4096*512 bf16)
  bf16* Wtq  = (bf16*)(w + 24 * MB);  // 0.5 MB (512*512 bf16, N-major)
  bf16* Wtk  = (bf16*)(w + 24 * MB + 512 * 1024);
  bf16* Wtv  = (bf16*)(w + 25 * MB);
  bf16* Wot  = (bf16*)(w + 25 * MB + 512 * 1024);

  prep_x_kernel<<<dim3(8192), 256, 0, stream>>>(x, xb);
  prep_w_kernel<<<dim3(1024), 256, 0, stream>>>(Wq, Wk, Wv, Wo,
                                                Wtq, Wtk, Wtv, Wot);
  qkv_kernel<<<dim3(32, 4, 3), 256, 0, stream>>>(xb, Wtq, Wtk, Wtv, bq, bk, bv,
                                                 Qb, Kb, Krot, Vt);
  attn_kernel<<<dim3(16, 8), 256, 0, stream>>>(Qb, Kb, Krot, Vt, Attn);
  oproj_kernel<<<dim3(32, 4), 256, 0, stream>>>(Attn, Wot, bo, out);
}
